// TestSiluMulGroupFp8QuantModel_34342558498900
// MI455X (gfx1250) — 8-bit matrix operand order repaired and checked by reading, compiled, not run on hardware
//
#include <hip/hip_runtime.h>
#include <math.h>

typedef __attribute__((ext_vector_type(16))) int   v16i;
typedef __attribute__((ext_vector_type(8)))  float v8f;

#define GROUP 128
#define FP8_MAX 448.0f

__device__ __forceinline__ unsigned f32_to_e4m3(float xf) {
  unsigned u = __float_as_uint(xf);
  unsigned s = (u >> 24) & 0x80u;
  float ax = __uint_as_float(u & 0x7fffffffu);
  if (!(ax < 480.0f)) return s | 0x7eu;
  unsigned a = __float_as_uint(ax);
  int e = (int)(a >> 23) - 127;
  unsigned q;
  if (e >= -6) {
    unsigned m    = (a >> 20) & 7u;
    unsigned rest = a & 0xfffffu;
    q = ((unsigned)(e + 7) << 3) | m;
    if (rest > 0x80000u || (rest == 0x80000u && (q & 1u))) q++;
    if (q > 0x7eu) q = 0x7eu;
  } else {
    q = (unsigned)(int)rintf(ax * 512.0f);
  }
  return s | q;
}

__global__ __launch_bounds__(256) void silu_quant_kernel(
    const float* __restrict__ x, unsigned char* __restrict__ yqs,
    float* __restrict__ a_scaleT, int T, int H) {
  const int HB  = H / GROUP;
  const int t   = blockIdx.x;
  const int tid = threadIdx.x;
  const float* gate = x + (size_t)t * (2 * H);
  const float* up   = gate + H;
  const int e0 = tid * 16;

  float y[16];
  float amax = 0.0f;
#pragma unroll
  for (int j = 0; j < 4; ++j) {
    float4 g = *(const float4*)(gate + e0 + j * 4);
    float4 u = *(const float4*)(up   + e0 + j * 4);
    float gv[4] = {g.x, g.y, g.z, g.w};
    float uv[4] = {u.x, u.y, u.z, u.w};
#pragma unroll
    for (int k = 0; k < 4; ++k) {
      float sg = gv[k] * (1.0f / (1.0f + __expf(-gv[k])));
      float yy = sg * uv[k];
      y[j * 4 + k] = yy;
      amax = fmaxf(amax, fabsf(yy));
    }
  }
#pragma unroll
  for (int off = 1; off < 8; off <<= 1)
    amax = fmaxf(amax, __shfl_xor(amax, off, 32));

  float scl = fmaxf(amax, 1e-12f) * (1.0f / FP8_MAX);
  float rs  = 1.0f / scl;

  int packed[4];
#pragma unroll
  for (int j = 0; j < 4; ++j) {
    unsigned b0 = f32_to_e4m3(y[j * 4 + 0] * rs);
    unsigned b1 = f32_to_e4m3(y[j * 4 + 1] * rs);
    unsigned b2 = f32_to_e4m3(y[j * 4 + 2] * rs);
    unsigned b3 = f32_to_e4m3(y[j * 4 + 3] * rs);
    packed[j] = (int)(b0 | (b1 << 8) | (b2 << 16) | (b3 << 24));
  }

  const int kb  = e0 >> 7;
  const int j16 = (e0 & 127) >> 4;
  const int tb  = t >> 4;
  const int r   = t & 15;
  unsigned char* tileB = yqs + ((size_t)tb * HB + kb) * (16 * GROUP);
  *(int4*)(tileB + (r + 16 * (j16 & 1)) * 64 + (j16 >> 1) * 16) = make_int4(packed[0], packed[1], packed[2], packed[3]);

  if ((tid & 7) == 0)
    a_scaleT[(size_t)(tid >> 3) * T + t] = scl;
}

__global__ __launch_bounds__(256) void wquant_kernel(
    const float* __restrict__ w, unsigned char* __restrict__ wqT, int H) {
  __shared__ unsigned char tile[32][33];
  const int n0 = blockIdx.x * 32;
  const int k0 = blockIdx.y * 32;
  const int tx = threadIdx.x, ty = threadIdx.y;
#pragma unroll
  for (int i = 0; i < 4; ++i) {
    int k = ty + i * 8;
    float v = w[(size_t)(k0 + k) * H + n0 + tx];
    tile[k][tx] = (unsigned char)f32_to_e4m3(v);
  }
  __syncthreads();
#pragma unroll
  for (int i = 0; i < 4; ++i) {
    int n = ty + i * 8;
    wqT[(size_t)(n0 + n) * H + k0 + tx] = tile[tx][n];
  }
}

union Frag { v16i v; int4 q[4]; };

__global__ __launch_bounds__(256) void gemm_fp8_kernel(
    const unsigned char* __restrict__ yqs, const unsigned char* __restrict__ wqT,
    const float* __restrict__ a_scaleT, const float* __restrict__ wscale,
    float* __restrict__ out, int T, int H) {
  const int HB   = H / GROUP;
  const int lane = threadIdx.x & 31;
  const int wave = threadIdx.x >> 5;
  const int waveM = wave >> 1;
  const int waveN = wave & 1;
  const int m0 = blockIdx.y * 128 + waveM * 32;
  const int n0 = blockIdx.x * 128 + waveN * 64;
  const int nb = blockIdx.x;
  const int half = lane >> 4;
  const int r    = lane & 15;

  const unsigned char* aBase[2];
#pragma unroll
  for (int mt = 0; mt < 2; ++mt)
    aBase[mt] = yqs + ((size_t)(m0 / 16 + mt) * HB) * (16 * GROUP) + lane * 64;

  const unsigned char* bBase[4];
#pragma unroll
  for (int nt = 0; nt < 4; ++nt)
    bBase[nt] = wqT + (size_t)(n0 + nt * 16 + r) * H + half * 16;

  const float* saBase[2];
#pragma unroll
  for (int mt = 0; mt < 2; ++mt)
    saBase[mt] = a_scaleT + (m0 + mt * 16 + half * 8);

  v8f acc[2][4];
#pragma unroll
  for (int mt = 0; mt < 2; ++mt)
#pragma unroll
    for (int nt = 0; nt < 4; ++nt)
      acc[mt][nt] = (v8f)0.0f;

  for (int kb = 0; kb < HB; ++kb) {
    Frag A[2], B[4];
#pragma unroll
    for (int mt = 0; mt < 2; ++mt) {
      const unsigned char* p = aBase[mt] + (size_t)kb * (16 * GROUP);
#pragma unroll
      for (int c = 0; c < 4; ++c)
        A[mt].q[c] = *(const int4*)(p + c * 16);
    }
#pragma unroll
    for (int nt = 0; nt < 4; ++nt) {
      const unsigned char* p = bBase[nt] + kb * GROUP;
#pragma unroll
      for (int c = 0; c < 4; ++c)
        B[nt].q[c] = *(const int4*)(p + c * 32);
    }

    const float sw = wscale[kb * HB + nb];
    float s[2][8];
#pragma unroll
    for (int mt = 0; mt < 2; ++mt) {
      const float* p = saBase[mt] + (size_t)kb * T;
      float4 lo = *(const float4*)(p);
      float4 hi = *(const float4*)(p + 4);
      s[mt][0] = lo.x * sw; s[mt][1] = lo.y * sw;
      s[mt][2] = lo.z * sw; s[mt][3] = lo.w * sw;
      s[mt][4] = hi.x * sw; s[mt][5] = hi.y * sw;
      s[mt][6] = hi.z * sw; s[mt][7] = hi.w * sw;
    }

#pragma unroll
    for (int mt = 0; mt < 2; ++mt)
#pragma unroll
      for (int nt = 0; nt < 4; ++nt) {
        v8f zero = (v8f)0.0f;
        v8f d = __builtin_amdgcn_wmma_f32_16x16x128_fp8_fp8(
            A[mt].v, B[nt].v, (short)0, zero, false, false);
#pragma unroll
        for (int v = 0; v < 8; ++v)
          acc[mt][nt][v] += s[mt][v] * d[v];
      }
  }

#pragma unroll
  for (int mt = 0; mt < 2; ++mt)
#pragma unroll
    for (int nt = 0; nt < 4; ++nt)
#pragma unroll
      for (int v = 0; v < 8; ++v)
        out[(size_t)(m0 + mt * 16 + half * 8 + v) * H + n0 + nt * 16 + r] =
            acc[mt][nt][v];
}

extern "C" void kernel_launch(void* const* d_in, const int* in_sizes, int n_in,
                              void* d_out, int out_size, void* d_ws, size_t ws_size,
                              hipStream_t stream) {
  const float* x      = (const float*)d_in[0];
  const float* w      = (const float*)d_in[1];
  const float* wscale = (const float*)d_in[2];
  float* out = (float*)d_out;

  int H = 1;
  while ((long long)H * H < (long long)in_sizes[1]) H <<= 1;
  int T = in_sizes[0] / (2 * H);

  unsigned char* ws  = (unsigned char*)d_ws;
  unsigned char* yqs = ws;
  unsigned char* wqT = ws + (size_t)T * H;
  float* a_scaleT    = (float*)(ws + (size_t)T * H + (size_t)H * H);

  silu_quant_kernel<<<T, 256, 0, stream>>>(x, yqs, a_scaleT, T, H);
  wquant_kernel<<<dim3(H / 32, H / 32), dim3(32, 8), 0, stream>>>(w, wqT, H);
  gemm_fp8_kernel<<<dim3(H / 128, T / 128), 256, 0, stream>>>(
      yqs, wqT, a_scaleT, wscale, out, T, H);
}
